// PerceiverResampler_80247168958807
// MI455X (gfx1250) — matrix operand order repaired and checked by reading, not compiled, not run on hardware
//
#include <hip/hip_runtime.h>

#define B_ 16
#define V_ 1024
#define D_ 1024
#define L_ 6
#define H_ 8
#define DH_ 64
#define I_ 512
#define NL_ 144
#define F_ 4096
#define EPSL 1e-5f

typedef __attribute__((ext_vector_type(16))) __bf16 v16bf;
typedef __attribute__((ext_vector_type(8)))  float  v8f;
typedef __attribute__((ext_vector_type(4)))  unsigned int v4u;
typedef __attribute__((ext_vector_type(4)))  int v4i;

union FragBF { v16bf v; v4u q[2]; };

__device__ __forceinline__ unsigned short f2bf(float f) {
  unsigned int u = __float_as_uint(f);
  u += 0x7FFFu + ((u >> 16) & 1u);
  return (unsigned short)(u >> 16);
}
__device__ __forceinline__ float bf2f(unsigned short h) {
  return __uint_as_float(((unsigned int)h) << 16);
}

// A-matrix 16x32 bf16 fragment: lane holds row = lane%16; half = lane/16.
// elems 0..7 -> K = half*8 + e ; elems 8..15 -> K = 16 + half*8 + (e-8)
__device__ __forceinline__ v16bf frag_a(const unsigned short* rowp, int half) {
  FragBF f;
  f.q[0] = *(const v4u*)(rowp + half * 8);
  f.q[1] = *(const v4u*)(rowp + 16 + half * 8);
  return f.v;
}
// B-matrix 32x16 bf16 fragment (column-major in LDS: [n][k]):
// lane holds col = lane%16; elems 0..7 -> K = half*8 + e ; elems 8..15 -> K = 16 + half*8 + (e-8)
__device__ __forceinline__ v16bf frag_b(const unsigned short* colp, int half) {
  FragBF f;
  f.q[0] = *(const v4u*)(colp + half * 8);
  f.q[1] = *(const v4u*)(colp + 16 + half * 8);
  return f.v;
}
__device__ __forceinline__ v8f wmma_bf16(v16bf a, v16bf b, v8f c) {
  return __builtin_amdgcn_wmma_f32_16x16x32_bf16(false, a, false, b, (short)0, c,
                                                 false, false);
}
__device__ __forceinline__ void wait_lds() {
#if __has_builtin(__builtin_amdgcn_s_wait_dscnt)
  __builtin_amdgcn_s_wait_dscnt(0);
#else
  asm volatile("s_wait_dscnt 0x0" ::: "memory");
#endif
}

// Async global->LDS staging (ASYNCcnt path). 16 bytes per lane per call.
#if __has_builtin(__builtin_amdgcn_global_load_async_to_lds_b128)
#define HAVE_ASYNC_LDS 1
#else
#define HAVE_ASYNC_LDS 0
#endif

__device__ __forceinline__ void async_copy16(const unsigned short* g,
                                             unsigned short* l) {
#if HAVE_ASYNC_LDS
  __builtin_amdgcn_global_load_async_to_lds_b128(
      (__attribute__((address_space(1))) v4i*)(void*)g,
      (__attribute__((address_space(3))) v4i*)(void*)l, 0, 0);
#else
  *(v4u*)l = *(const v4u*)g;
#endif
}
__device__ __forceinline__ void wait_async_all() {
#if HAVE_ASYNC_LDS
#if __has_builtin(__builtin_amdgcn_s_wait_asynccnt)
  __builtin_amdgcn_s_wait_asynccnt(0);
#else
  asm volatile("s_wait_asynccnt 0x0" ::: "memory");
#endif
#endif
}

// ---------------------------------------------------------------- converts
__global__ __launch_bounds__(256) void cvt_bf16_kernel(
    const float* __restrict__ X, unsigned short* __restrict__ Y, int n) {
  int i = blockIdx.x * 256 + threadIdx.x;
  if (i < n) Y[i] = f2bf(X[i]);
}

__global__ __launch_bounds__(256) void lat_init_kernel(
    const float* __restrict__ Lt, float* __restrict__ lat) {
  int i = blockIdx.x * 256 + threadIdx.x;
  if (i < B_ * NL_ * D_) lat[i] = Lt[i % (NL_ * D_)];
}

// ---------------------------------------------------------------- layernorm
template <bool OUT_BF16>
__global__ __launch_bounds__(256) void ln_kernel(const float* __restrict__ X,
                                                 const float* __restrict__ g,
                                                 const float* __restrict__ b,
                                                 void* __restrict__ Y) {
  const int row = blockIdx.x;
  const int t = threadIdx.x;
  const float* xr = X + (size_t)row * D_;
  float v[4];
  float s = 0.f;
#pragma unroll
  for (int i = 0; i < 4; i++) { v[i] = xr[t + i * 256]; s += v[i]; }
  __shared__ float red[8];
#pragma unroll
  for (int o = 16; o >= 1; o >>= 1) s += __shfl_xor(s, o, 32);
  if ((t & 31) == 0) red[t >> 5] = s;
  __syncthreads();
  float tot = 0.f;
#pragma unroll
  for (int i = 0; i < 8; i++) tot += red[i];
  const float mean = tot * (1.f / D_);
  float s2 = 0.f;
#pragma unroll
  for (int i = 0; i < 4; i++) { float d = v[i] - mean; s2 += d * d; }
#pragma unroll
  for (int o = 16; o >= 1; o >>= 1) s2 += __shfl_xor(s2, o, 32);
  __syncthreads();
  if ((t & 31) == 0) red[t >> 5] = s2;
  __syncthreads();
  float t2 = 0.f;
#pragma unroll
  for (int i = 0; i < 8; i++) t2 += red[i];
  const float rstd = rsqrtf(t2 * (1.f / D_) + EPSL);
#pragma unroll
  for (int i = 0; i < 4; i++) {
    const int col = t + i * 256;
    const float y = (v[i] - mean) * rstd * g[col] + b[col];
    if (OUT_BF16)
      ((unsigned short*)Y)[(size_t)row * D_ + col] = f2bf(y);
    else
      ((float*)Y)[(size_t)row * D_ + col] = y;
  }
}

// ---------------------------------------------------------------- WMMA GEMM
// C(MxN) = A(MxK, bf16) @ B(KxN, bf16)  [+ R]  [gelu]
// EPI: 0 = none, 1 = +R residual (f32), 2 = exact GELU
// Double-buffered LDS; A tile staged with async global->LDS b128 copies,
// B tile software-pipelined through VGPRs with a 16-bit transpose store.
#define GTM 128
#define GTN 128
#define GTK 32
#define SLA 40
#define SLB 40

template <int EPI, bool OBF>
__global__ __launch_bounds__(256) void gemm_bf16_kernel(
    const unsigned short* __restrict__ A, const unsigned short* __restrict__ Bm,
    void* __restrict__ C, const float* __restrict__ R, int M, int N, int K) {
  __shared__ __align__(16) unsigned short As[2][GTM * SLA];
  __shared__ __align__(16) unsigned short Bs[2][GTN * SLB];
  const int tid = threadIdx.x;
  const int lane = tid & 31;
  const int wid = tid >> 5;
  const int half = lane >> 4;
  const int ln16 = lane & 15;
  const int tm0 = blockIdx.y * GTM;
  const int tn0 = blockIdx.x * GTN;
  const int wm = (wid >> 1) * 32;  // 0,32,64,96
  const int wn = (wid & 1) * 64;   // 0,64

  v8f acc[2][4];
#pragma unroll
  for (int i = 0; i < 2; i++)
#pragma unroll
    for (int j = 0; j < 4; j++)
#pragma unroll
      for (int e = 0; e < 8; e++) acc[i][j][e] = 0.f;

  const int arow = tid >> 1;
  const int acol = (tid & 1) * 16;
  const int bk = tid >> 3;
  const int bn = (tid & 7) * 16;

  auto stage_a = [&](int k0, int p) {
    const unsigned short* src = A + (size_t)(tm0 + arow) * K + k0 + acol;
    unsigned short* dst = &As[p][arow * SLA + acol];
    async_copy16(src, dst);
    async_copy16(src + 8, dst + 8);
  };
  auto load_b_regs = [&](int k0, v4u& b0, v4u& b1) {
    const v4u* bsrc = (const v4u*)(Bm + (size_t)(k0 + bk) * N + tn0 + bn);
    b0 = bsrc[0];
    b1 = bsrc[1];
  };
  auto store_b_lds = [&](const v4u& b0, const v4u& b1, int p) {
#pragma unroll
    for (int i = 0; i < 4; i++) {
      const unsigned int w0 = b0[i], w1 = b1[i];
      Bs[p][(bn + 2 * i + 0) * SLB + bk] = (unsigned short)(w0 & 0xffffu);
      Bs[p][(bn + 2 * i + 1) * SLB + bk] = (unsigned short)(w0 >> 16);
      Bs[p][(bn + 8 + 2 * i + 0) * SLB + bk] = (unsigned short)(w1 & 0xffffu);
      Bs[p][(bn + 8 + 2 * i + 1) * SLB + bk] = (unsigned short)(w1 >> 16);
    }
  };

  // prologue: stage tile 0 into buffer 0
  {
    v4u b0, b1;
    load_b_regs(0, b0, b1);
    stage_a(0, 0);
    store_b_lds(b0, b1, 0);
    wait_async_all();
  }
  __syncthreads();

  int p = 0;
  for (int k0 = 0; k0 < K; k0 += GTK, p ^= 1) {
    const int nk = k0 + GTK;
    const bool more = nk < K;
    v4u nb0, nb1;
    if (more) {
      stage_a(nk, p ^ 1);          // async A -> other buffer
      load_b_regs(nk, nb0, nb1);   // B next K-step -> VGPRs
    }
    if (nk + GTK < K) {
      __builtin_prefetch(Bm + (size_t)(nk + GTK + bk) * N + tn0 + bn, 0, 1);
      __builtin_prefetch(A + (size_t)(tm0 + arow) * K + nk + GTK + acol, 0, 1);
    }
    // compute on current buffer while copies are in flight
    const v16bf af0 = frag_a(&As[p][(wm + ln16) * SLA], half);
    const v16bf af1 = frag_a(&As[p][(wm + 16 + ln16) * SLA], half);
#pragma unroll
    for (int j = 0; j < 4; j++) {
      const v16bf bfb = frag_b(&Bs[p][(wn + j * 16 + ln16) * SLB], half);
      acc[0][j] = wmma_bf16(af0, bfb, acc[0][j]);
      acc[1][j] = wmma_bf16(af1, bfb, acc[1][j]);
    }
    if (more) store_b_lds(nb0, nb1, p ^ 1);
    wait_async_all();
    __syncthreads();
  }

#pragma unroll
  for (int i = 0; i < 2; i++)
#pragma unroll
    for (int j = 0; j < 4; j++)
#pragma unroll
      for (int r = 0; r < 8; r++) {
        const int grow = tm0 + wm + i * 16 + r + 8 * half;
        const int gcol = tn0 + wn + j * 16 + ln16;
        float val = acc[i][j][r];
        if (EPI == 1) val += R[(size_t)grow * N + gcol];
        if (EPI == 2) val = 0.5f * val * (1.f + erff(val * 0.70710678118f));
        if (OBF)
          ((unsigned short*)C)[(size_t)grow * N + gcol] = f2bf(val);
        else
          ((float*)C)[(size_t)grow * N + gcol] = val;
      }
}

// ---------------------------------------------------------------- attention
// Flash-style: one block per (b,h); 10 waves; wave w owns latent rows 16w..16w+15
// (rows >=144 are zero padding). Streams keys in chunks of 64.
#define AQL 72   // q_s row stride (64 + 8)
#define AKL 72   // k_s row stride
#define AVL 72   // vt_s row stride
#define APL 72   // p_s row stride

__global__ __launch_bounds__(320) void attn_kernel(
    const unsigned short* __restrict__ Q,   // (B*NL, I) bf16
    const unsigned short* __restrict__ KV,  // (B*V, 2I) bf16
    unsigned short* __restrict__ O) {       // (B*NL, I) bf16
  __shared__ __align__(16) unsigned short q_s[160 * AQL];
  __shared__ __align__(16) unsigned short k_s[64 * AKL];    // [j][d]
  __shared__ __align__(16) unsigned short vt_s[64 * AVL];   // [d][j]
  __shared__ __align__(16) unsigned short p_s[10 * 16 * APL];

  const int bh = blockIdx.x;
  const int b = bh >> 3;
  const int h = bh & 7;
  const int tid = threadIdx.x;
  const int wid = tid >> 5;
  const int lane = tid & 31;
  const int half = lane >> 4;
  const int ln16 = lane & 15;

  // load q (scaled by DH^-0.5 = 0.125), pad rows >= NL with zero
  for (int idx = tid; idx < 160 * 64; idx += 320) {
    const int r = idx >> 6, d = idx & 63;
    float val = 0.f;
    if (r < NL_) val = 0.125f * bf2f(Q[((size_t)(b * NL_ + r)) * I_ + h * DH_ + d]);
    q_s[r * AQL + d] = f2bf(val);
  }

  v8f o_acc[4];
#pragma unroll
  for (int ot = 0; ot < 4; ot++)
#pragma unroll
    for (int e = 0; e < 8; e++) o_acc[ot][e] = 0.f;
  float m_row[8], l_row[8];
#pragma unroll
  for (int r = 0; r < 8; r++) { m_row[r] = -1e30f; l_row[r] = 0.f; }

  const unsigned short* qrow = &q_s[(wid * 16 + ln16) * AQL];
  unsigned short* pw = &p_s[wid * 16 * APL];

  for (int c = 0; c < V_ / 64; ++c) {
    __syncthreads();  // protect k_s/vt_s (and q_s on first iter)
    for (int idx = tid; idx < 64 * 64; idx += 320) {
      const int j = idx >> 6, d = idx & 63;
      const size_t base = ((size_t)(b * V_ + c * 64 + j)) * (2 * I_) + h * DH_;
      k_s[j * AKL + d] = KV[base + d];
      vt_s[d * AVL + j] = KV[base + I_ + d];
    }
    __syncthreads();

    // S = q @ k^T  (16 x 64 per wave)
    float p[4][8];
#pragma unroll
    for (int nt = 0; nt < 4; ++nt) {
      v8f sacc;
#pragma unroll
      for (int e = 0; e < 8; e++) sacc[e] = 0.f;
#pragma unroll
      for (int ks = 0; ks < 2; ++ks) {
        const v16bf a = frag_a(qrow + ks * 32, half);
        const v16bf bb = frag_b(&k_s[(nt * 16 + ln16) * AKL + ks * 32], half);
        sacc = wmma_bf16(a, bb, sacc);
      }
#pragma unroll
      for (int r = 0; r < 8; ++r) p[nt][r] = sacc[r];
    }
    // row max (rows live on 16-lane halves; xor masks 1..8 stay in-half)
    float mx[8];
#pragma unroll
    for (int r = 0; r < 8; ++r) {
      float m = p[0][r];
#pragma unroll
      for (int nt = 1; nt < 4; ++nt) m = fmaxf(m, p[nt][r]);
      mx[r] = m;
    }
#pragma unroll
    for (int o = 8; o >= 1; o >>= 1)
#pragma unroll
      for (int r = 0; r < 8; ++r) mx[r] = fmaxf(mx[r], __shfl_xor(mx[r], o, 32));
    float alpha[8];
#pragma unroll
    for (int r = 0; r < 8; ++r) {
      const float mn = fmaxf(m_row[r], mx[r]);
      alpha[r] = __expf(m_row[r] - mn);
      m_row[r] = mn;
    }
    float ls[8];
#pragma unroll
    for (int r = 0; r < 8; ++r) ls[r] = 0.f;
#pragma unroll
    for (int nt = 0; nt < 4; ++nt)
#pragma unroll
      for (int r = 0; r < 8; ++r) {
        const float e = __expf(p[nt][r] - m_row[r]);
        p[nt][r] = e;
        ls[r] += e;
      }
#pragma unroll
    for (int o = 8; o >= 1; o >>= 1)
#pragma unroll
      for (int r = 0; r < 8; ++r) ls[r] += __shfl_xor(ls[r], o, 32);
#pragma unroll
    for (int r = 0; r < 8; ++r) l_row[r] = l_row[r] * alpha[r] + ls[r];
#pragma unroll
    for (int ot = 0; ot < 4; ++ot)
#pragma unroll
      for (int r = 0; r < 8; ++r) o_acc[ot][r] *= alpha[r];
    // stage P (bf16) for use as A-matrix of O += P @ V (per-wave buffer)
#pragma unroll
    for (int nt = 0; nt < 4; ++nt)
#pragma unroll
      for (int r = 0; r < 8; ++r)
        pw[(r + 8 * half) * APL + nt * 16 + ln16] = f2bf(p[nt][r]);
    wait_lds();
#pragma unroll
    for (int ot = 0; ot < 4; ++ot)
#pragma unroll
      for (int ks = 0; ks < 2; ++ks) {
        const v16bf a = frag_a(pw + ln16 * APL + ks * 32, half);
        const v16bf bb = frag_b(&vt_s[(ot * 16 + ln16) * AVL + ks * 32], half);
        o_acc[ot] = wmma_bf16(a, bb, o_acc[ot]);
      }
  }
#pragma unroll
  for (int ot = 0; ot < 4; ++ot)
#pragma unroll
    for (int r = 0; r < 8; ++r) {
      const int row = wid * 16 + r + 8 * half;
      if (row < NL_) {
        const float v = o_acc[ot][r] / l_row[r];
        O[((size_t)(b * NL_ + row)) * I_ + h * DH_ + ot * 16 + ln16] = f2bf(v);
      }
    }
}

// ---------------------------------------------------------------- launch
extern "C" void kernel_launch(void* const* d_in, const int* in_sizes, int n_in,
                              void* d_out, int out_size, void* d_ws,
                              size_t ws_size, hipStream_t stream) {
  (void)in_sizes; (void)n_in; (void)out_size; (void)ws_size;
  const float* x       = (const float*)d_in[0];
  const float* latents = (const float*)d_in[1];
  const float* Wq      = (const float*)d_in[2];
  const float* Wkv     = (const float*)d_in[3];
  const float* Wo      = (const float*)d_in[4];
  const float* W1      = (const float*)d_in[5];
  const float* W2      = (const float*)d_in[6];
  const float* ln_m_g  = (const float*)d_in[7];
  const float* ln_m_b  = (const float*)d_in[8];
  const float* ln_l_g  = (const float*)d_in[9];
  const float* ln_l_b  = (const float*)d_in[10];
  const float* ff_g    = (const float*)d_in[11];
  const float* ff_b    = (const float*)d_in[12];
  const float* norm_g  = (const float*)d_in[13];
  const float* norm_b  = (const float*)d_in[14];

  char* wsp = (char*)d_ws;
  auto take = [&](size_t bytes) -> char* {
    char* p = wsp;
    wsp += (bytes + 255) & ~(size_t)255;
    return p;
  };
  unsigned short* wq_bf  = (unsigned short*)take((size_t)L_ * D_ * I_ * 2);
  unsigned short* wkv_bf = (unsigned short*)take((size_t)L_ * D_ * 2 * I_ * 2);
  unsigned short* wo_bf  = (unsigned short*)take((size_t)L_ * I_ * D_ * 2);
  unsigned short* w1_bf  = (unsigned short*)take((size_t)L_ * D_ * F_ * 2);
  unsigned short* w2_bf  = (unsigned short*)take((size_t)L_ * F_ * D_ * 2);
  unsigned short* xm_bf  = (unsigned short*)take((size_t)B_ * V_ * D_ * 2);
  unsigned short* kv_bf  = (unsigned short*)take((size_t)B_ * V_ * 2 * I_ * 2);
  float*          lat    = (float*)take((size_t)B_ * NL_ * D_ * 4);
  unsigned short* lm_bf  = (unsigned short*)take((size_t)B_ * NL_ * D_ * 2);
  unsigned short* q_bf   = (unsigned short*)take((size_t)B_ * NL_ * I_ * 2);
  unsigned short* ao_bf  = (unsigned short*)take((size_t)B_ * NL_ * I_ * 2);
  unsigned short* lnf_bf = (unsigned short*)take((size_t)B_ * NL_ * D_ * 2);
  unsigned short* h_bf   = (unsigned short*)take((size_t)B_ * NL_ * F_ * 2);

  auto cvt = [&](const float* src, unsigned short* dst, size_t n) {
    cvt_bf16_kernel<<<dim3((unsigned)((n + 255) / 256)), dim3(256), 0, stream>>>(
        src, dst, (int)n);
  };
  cvt(Wq, wq_bf, (size_t)L_ * D_ * I_);
  cvt(Wkv, wkv_bf, (size_t)L_ * D_ * 2 * I_);
  cvt(Wo, wo_bf, (size_t)L_ * I_ * D_);
  cvt(W1, w1_bf, (size_t)L_ * D_ * F_);
  cvt(W2, w2_bf, (size_t)L_ * F_ * D_);

  lat_init_kernel<<<dim3((B_ * NL_ * D_ + 255) / 256), dim3(256), 0, stream>>>(
      latents, lat);

  const int MR = B_ * NL_;  // 2304
  for (int l = 0; l < L_; ++l) {
    // xm = LN(x) -> bf16 ; kv = xm @ Wkv[l]
    ln_kernel<true><<<dim3(B_ * V_), dim3(256), 0, stream>>>(
        x, ln_m_g + l * D_, ln_m_b + l * D_, xm_bf);
    gemm_bf16_kernel<0, true>
        <<<dim3((2 * I_) / GTN, (B_ * V_) / GTM), dim3(256), 0, stream>>>(
            xm_bf, wkv_bf + (size_t)l * D_ * 2 * I_, kv_bf, nullptr, B_ * V_,
            2 * I_, D_);
    // lm = LN(lat) -> bf16 ; q = lm @ Wq[l]
    ln_kernel<true><<<dim3(MR), dim3(256), 0, stream>>>(
        lat, ln_l_g + l * D_, ln_l_b + l * D_, lm_bf);
    gemm_bf16_kernel<0, true>
        <<<dim3(I_ / GTN, MR / GTM), dim3(256), 0, stream>>>(
            lm_bf, wq_bf + (size_t)l * D_ * I_, q_bf, nullptr, MR, I_, D_);
    // attention
    attn_kernel<<<dim3(B_ * H_), dim3(320), 0, stream>>>(q_bf, kv_bf, ao_bf);
    // lat = attn_out @ Wo[l] + lat
    gemm_bf16_kernel<1, false>
        <<<dim3(D_ / GTN, MR / GTM), dim3(256), 0, stream>>>(
            ao_bf, wo_bf + (size_t)l * I_ * D_, lat, lat, MR, D_, I_);
    // FF: h = gelu(LN(lat) @ W1) ; lat = h @ W2 + lat
    ln_kernel<true><<<dim3(MR), dim3(256), 0, stream>>>(
        lat, ff_g + l * D_, ff_b + l * D_, lnf_bf);
    gemm_bf16_kernel<2, true>
        <<<dim3(F_ / GTN, MR / GTM), dim3(256), 0, stream>>>(
            lnf_bf, w1_bf + (size_t)l * D_ * F_, h_bf, nullptr, MR, F_, D_);
    gemm_bf16_kernel<1, false>
        <<<dim3(D_ / GTN, MR / GTM), dim3(256), 0, stream>>>(
            h_bf, w2_bf + (size_t)l * F_ * D_, lat, lat, MR, D_, F_);
  }
  ln_kernel<false><<<dim3(MR), dim3(256), 0, stream>>>(lat, norm_g, norm_b,
                                                       d_out);
}
